// VisionTransformer_31688268710300
// MI455X (gfx1250) — 16-bit and 8-bit matrix operand order repaired and checked by reading, compiled, not run on hardware
//
#include <hip/hip_runtime.h>
#include <hip/hip_fp16.h>

typedef __attribute__((ext_vector_type(8)))  int      v8i;
typedef __attribute__((ext_vector_type(8)))  float    v8f;
typedef __attribute__((ext_vector_type(16))) _Float16 v16h;

#define BATCH   32
#define NTOK    197
#define TOK     (BATCH*NTOK)
#define DIMM    768
#define HEADS   12
#define HD      64
#define PTOK    196
#define CONVTOK (BATCH*PTOK)
#define ZBH     (BATCH*HEADS)
#define KPAD    224

static constexpr unsigned long long alup(unsigned long long x){ return (x+255ULL) & ~255ULL; }
static constexpr unsigned long long O_SC    = 0;
static constexpr unsigned long long O_IM2   = alup(O_SC    + 8192);
static constexpr unsigned long long O_WQKV  = alup(O_IM2   + 6272ULL*768);
static constexpr unsigned long long O_WPROJ = alup(O_WQKV  + 2304ULL*768);
static constexpr unsigned long long O_WFC1  = alup(O_WPROJ + 768ULL*768);
static constexpr unsigned long long O_WFC2  = alup(O_WFC1  + 3072ULL*768);
static constexpr unsigned long long O_WHEAD = alup(O_WFC2  + 768ULL*3072);
static constexpr unsigned long long O_WCONV = alup(O_WHEAD + 1000ULL*768);
static constexpr unsigned long long O_CLSI8 = alup(O_WCONV + 768ULL*768);
static constexpr unsigned long long O_H     = alup(O_CLSI8 + 32ULL*768);
static constexpr unsigned long long O_A     = alup(O_H     + 6304ULL*768*4);
static constexpr unsigned long long O_AI8   = alup(O_A     + 6304ULL*768*4);
static constexpr unsigned long long O_QKV   = alup(O_AI8   + 6304ULL*768);
static constexpr unsigned long long O_QKVI8 = alup(O_QKV   + 6304ULL*2304*4);
static constexpr unsigned long long O_ATTN  = alup(O_QKVI8 + 6304ULL*2304);
static constexpr unsigned long long O_ATTNH = alup(O_ATTN  + 384ULL*197*197*4);
static constexpr unsigned long long O_OBUF  = alup(O_ATTNH + 384ULL*197*224*2);
static constexpr unsigned long long O_M1    = alup(O_OBUF  + 6304ULL*768*4);
static constexpr unsigned long long O_M1I8  = alup(O_M1    + 6304ULL*3072*4);

__device__ __forceinline__ void vit_async_b128(unsigned ldsAddr, const void* gaddr){
  asm volatile("global_load_async_to_lds_b128 %0, %1, off"
               :: "v"(ldsAddr), "v"((unsigned long long)gaddr) : "memory");
}
__device__ __forceinline__ void vit_wait_async4(){ asm volatile("s_wait_asynccnt 0x4" ::: "memory"); }
__device__ __forceinline__ void vit_wait_async0(){ asm volatile("s_wait_asynccnt 0x0" ::: "memory"); }

__global__ void vit_zero_f32(float* p, int n){
  int i = blockIdx.x*blockDim.x + threadIdx.x;
  if (i < n) p[i] = 0.f;
}

__global__ __launch_bounds__(256) void vit_absmax(const float* __restrict__ x, long long n, float* __restrict__ slot){
  __shared__ float red[256];
  float m = 0.f;
  long long stride = (long long)gridDim.x*blockDim.x;
  for (long long i = (long long)blockIdx.x*blockDim.x + threadIdx.x; i < n; i += stride){
    if (i + stride < n) __builtin_prefetch(&x[i + stride], 0, 0);
    m = fmaxf(m, fabsf(x[i]));
  }
  red[threadIdx.x] = m; __syncthreads();
  for (int s = 128; s > 0; s >>= 1){
    if (threadIdx.x < s) red[threadIdx.x] = fmaxf(red[threadIdx.x], red[threadIdx.x+s]);
    __syncthreads();
  }
  if (threadIdx.x == 0) atomicMax((unsigned int*)slot, __float_as_uint(red[0]));
}

__global__ void vit_fin_scale(float* slot, float nlev){
  slot[1] = fmaxf(slot[0]/nlev, 1e-8f);
}

__global__ __launch_bounds__(256) void vit_snap(float* __restrict__ x, long long n,
                                                const float* __restrict__ sp, float nlev){
  float s = sp[0];
  for (long long i = (long long)blockIdx.x*blockDim.x + threadIdx.x; i < n;
       i += (long long)gridDim.x*blockDim.x){
    float q = rintf(x[i]/s);
    q = fminf(fmaxf(q, -nlev), nlev);
    x[i] = q*s;
  }
}

__global__ __launch_bounds__(256) void vit_quant_i8(const float* __restrict__ x, signed char* __restrict__ q,
                                                    long long n, const float* __restrict__ sp){
  float s = sp[0];
  for (long long i = (long long)blockIdx.x*blockDim.x + threadIdx.x; i < n;
       i += (long long)gridDim.x*blockDim.x){
    float v = rintf(x[i]/s);
    v = fminf(fmaxf(v, -127.f), 127.f);
    q[i] = (signed char)v;
  }
}

__global__ __launch_bounds__(256) void vit_add(float* __restrict__ h, const float* __restrict__ o, long long n){
  for (long long i = (long long)blockIdx.x*blockDim.x + threadIdx.x; i < n;
       i += (long long)gridDim.x*blockDim.x)
    h[i] += o[i];
}

__global__ __launch_bounds__(256) void vit_gelu(float* __restrict__ x, long long n){
  for (long long i = (long long)blockIdx.x*blockDim.x + threadIdx.x; i < n;
       i += (long long)gridDim.x*blockDim.x){
    float v = x[i];
    float c = 0.7978845608028654f*(v + 0.044715f*v*v*v);
    x[i] = 0.5f*v*(1.f + tanhf(c));
  }
}

__global__ __launch_bounds__(256) void vit_sumabs(const float* __restrict__ x, long long n, float* __restrict__ slot){
  __shared__ float red[256];
  float s = 0.f;
  for (long long i = (long long)blockIdx.x*blockDim.x + threadIdx.x; i < n;
       i += (long long)gridDim.x*blockDim.x)
    s += fabsf(x[i]);
  red[threadIdx.x] = s; __syncthreads();
  for (int k = 128; k > 0; k >>= 1){
    if (threadIdx.x < k) red[threadIdx.x] += red[threadIdx.x+k];
    __syncthreads();
  }
  if (threadIdx.x == 0) atomicAdd(slot, red[0]);
}

__global__ void vit_ter_thr(float* slot, float nelem){ slot[1] = 0.7f * slot[0] / nelem; }

__global__ __launch_bounds__(256) void vit_ter_masked(const float* __restrict__ x, long long n, float* __restrict__ slot){
  __shared__ float ra[256];
  __shared__ float rc[256];
  float thr = slot[1];
  float sa = 0.f, cnt = 0.f;
  for (long long i = (long long)blockIdx.x*blockDim.x + threadIdx.x; i < n;
       i += (long long)gridDim.x*blockDim.x){
    float a = fabsf(x[i]);
    if (a > thr){ sa += a; cnt += 1.f; }
  }
  ra[threadIdx.x] = sa; rc[threadIdx.x] = cnt; __syncthreads();
  for (int k = 128; k > 0; k >>= 1){
    if (threadIdx.x < k){ ra[threadIdx.x] += ra[threadIdx.x+k]; rc[threadIdx.x] += rc[threadIdx.x+k]; }
    __syncthreads();
  }
  if (threadIdx.x == 0){ atomicAdd(slot+2, ra[0]); atomicAdd(slot+3, rc[0]); }
}

__global__ void vit_ter_alpha(float* slot){ slot[4] = slot[2] / fmaxf(slot[3], 1.f); }

__global__ __launch_bounds__(256) void vit_ter_write(const float* __restrict__ x, signed char* __restrict__ q,
                                                     long long n, const float* __restrict__ slot){
  float thr = slot[1];
  for (long long i = (long long)blockIdx.x*blockDim.x + threadIdx.x; i < n;
       i += (long long)gridDim.x*blockDim.x){
    float v = x[i];
    q[i] = (fabsf(v) > thr) ? (signed char)(v > 0.f ? 1 : -1) : (signed char)0;
  }
}

__global__ __launch_bounds__(256) void vit_im2col_q(const float* __restrict__ x, signed char* __restrict__ q,
                                                    const float* __restrict__ sp){
  float s = sp[0];
  long long total = (long long)CONVTOK * DIMM;
  for (long long i = (long long)blockIdx.x*blockDim.x + threadIdx.x; i < total;
       i += (long long)gridDim.x*blockDim.x){
    int m = (int)(i / DIMM), k = (int)(i % DIMM);
    int b = m / PTOK, t = m % PTOK, py = t / 14, px = t % 14;
    int c = k >> 8, r = (k >> 4) & 15, cc = k & 15;
    float v = x[(((long long)(b*3 + c)*224 + py*16 + r)*224 + px*16 + cc)];
    float qq = rintf(v/s);
    qq = fminf(fmaxf(qq, -127.f), 127.f);
    q[i] = (signed char)qq;
  }
}

__global__ __launch_bounds__(256) void vit_build_h(float* __restrict__ H, const float* __restrict__ cls,
                                                   const float* __restrict__ pos, const float* __restrict__ conv,
                                                   const float* __restrict__ spPos){
  float s = spPos[0];
  long long total = (long long)TOK * DIMM;
  for (long long i = (long long)blockIdx.x*blockDim.x + threadIdx.x; i < total;
       i += (long long)gridDim.x*blockDim.x){
    int b = (int)(i / ((long long)NTOK*DIMM));
    int rem = (int)(i % ((long long)NTOK*DIMM));
    int n = rem / DIMM, c = rem % DIMM;
    float p = pos[n*DIMM + c];
    float pq = fminf(fmaxf(rintf(p/s), -127.f), 127.f) * s;
    float base = (n == 0) ? cls[c] : conv[((long long)(b*PTOK + (n-1)))*DIMM + c];
    H[i] = base + pq;
  }
}

__global__ __launch_bounds__(256) void vit_ln(const float* __restrict__ x, const float* __restrict__ w,
                                              const float* __restrict__ b, float* __restrict__ y, int rows){
  int wave = threadIdx.x >> 5, lane = threadIdx.x & 31;
  long long row = (long long)blockIdx.x*8 + wave;
  if (row >= rows) return;
  const float* xr = x + row*DIMM;
  float vals[24];
  float sum = 0.f;
#pragma unroll
  for (int j = 0; j < 24; ++j){ vals[j] = xr[lane + j*32]; sum += vals[j]; }
  for (int o = 16; o > 0; o >>= 1) sum += __shfl_xor(sum, o, 32);
  float mu = sum * (1.f/768.f);
  float var = 0.f;
#pragma unroll
  for (int j = 0; j < 24; ++j){ float d = vals[j]-mu; var += d*d; }
  for (int o = 16; o > 0; o >>= 1) var += __shfl_xor(var, o, 32);
  var *= (1.f/768.f);
  float r = rsqrtf(var + 1e-6f);
  float* yr = y + row*DIMM;
#pragma unroll
  for (int j = 0; j < 24; ++j){
    int c = lane + j*32;
    yr[c] = (vals[j]-mu)*r*w[c] + b[c];
  }
}

__global__ __launch_bounds__(256) void vit_softmax(const float* __restrict__ attn, __half* __restrict__ outh,
                                                   const float* __restrict__ sp){
  int wave = threadIdx.x >> 5, lane = threadIdx.x & 31;
  long long row = (long long)blockIdx.x*8 + wave;
  if (row >= (long long)ZBH*NTOK) return;
  const float* ar = attn + row*NTOK;
  __half* orow = outh + row*KPAD;
  float s = sp[0];
  float v[7];
  float mx = -1e30f;
#pragma unroll
  for (int j = 0; j < 7; ++j){
    int c = lane + j*32;
    float t = -1e30f;
    if (c < NTOK){
      float q = rintf(ar[c]/s);
      q = fminf(fmaxf(q, -127.f), 127.f);
      t = q*s;
    }
    v[j] = t; mx = fmaxf(mx, t);
  }
  for (int o = 16; o > 0; o >>= 1) mx = fmaxf(mx, __shfl_xor(mx, o, 32));
  float sum = 0.f;
#pragma unroll
  for (int j = 0; j < 7; ++j){
    int c = lane + j*32;
    float e = 0.f;
    if (c < NTOK) e = __expf(v[j]-mx);
    v[j] = e; sum += e;
  }
  for (int o = 16; o > 0; o >>= 1) sum += __shfl_xor(sum, o, 32);
  float inv = 1.f/sum;
#pragma unroll
  for (int j = 0; j < 7; ++j){
    int c = lane + j*32;
    float p = (c < NTOK) ? rintf(v[j]*inv*65535.f)*(1.f/65535.f) : 0.f;
    orow[c] = __float2half(p);
  }
}

__global__ void vit_absmax_cls(const float* __restrict__ a, float* __restrict__ slot){
  __shared__ float red[256];
  int i = blockIdx.x*blockDim.x + threadIdx.x;
  float m = 0.f;
  if (i < BATCH*DIMM){
    int b = i / DIMM, c = i % DIMM;
    m = fabsf(a[(long long)b*NTOK*DIMM + c]);
  }
  red[threadIdx.x] = m; __syncthreads();
  for (int s = 128; s > 0; s >>= 1){
    if (threadIdx.x < s) red[threadIdx.x] = fmaxf(red[threadIdx.x], red[threadIdx.x+s]);
    __syncthreads();
  }
  if (threadIdx.x == 0) atomicMax((unsigned int*)slot, __float_as_uint(red[0]));
}

__global__ void vit_quant_cls(const float* __restrict__ a, signed char* __restrict__ q,
                              const float* __restrict__ sp){
  int i = blockIdx.x*blockDim.x + threadIdx.x;
  if (i >= BATCH*DIMM) return;
  int b = i / DIMM, c = i % DIMM;
  float s = sp[0];
  float v = rintf(a[(long long)b*NTOK*DIMM + c]/s);
  v = fminf(fmaxf(v, -127.f), 127.f);
  q[i] = (signed char)v;
}

#define LDT    80
#define TILE_B (128*LDT)

__device__ __forceinline__ void vit_mma8(const signed char* As, const signed char* Bs,
                                         int wm, int wn, int hi, int mr, v8i acc[2][4]){
  v8i af[2];
  v8i bf[4];
#pragma unroll
  for (int fm = 0; fm < 2; ++fm){
    const signed char* ap = As + (wm*32 + fm*16 + mr)*LDT + hi*8;
#pragma unroll
    for (int i = 0; i < 8; ++i){
      int off = ((i & 1) << 2) + ((i & 2) << 3) + ((i & 4) << 3);
      af[fm][i] = *(const int*)(ap + off);
    }
  }
#pragma unroll
  for (int fn = 0; fn < 4; ++fn){
    const signed char* bp = Bs + (wn*64 + fn*16 + mr)*LDT + hi*8;
#pragma unroll
    for (int i = 0; i < 8; ++i){
      int off = ((i & 1) << 2) + ((i & 2) << 3) + ((i & 4) << 3);
      bf[fn][i] = *(const int*)(bp + off);
    }
  }
#pragma unroll
  for (int fm = 0; fm < 2; ++fm)
#pragma unroll
    for (int fn = 0; fn < 4; ++fn)
      acc[fm][fn] = __builtin_amdgcn_wmma_i32_16x16x64_iu8(true, af[fm], true, bf[fn],
                                                           acc[fm][fn], false, false);
}

__global__ __launch_bounds__(256)
void vit_gemm_i8(const signed char* __restrict__ A, int lda, long long aOut, long long aIn,
                 const signed char* __restrict__ B, int ldb, long long bOut, long long bIn,
                 float* __restrict__ C, int ldc, long long cOut, long long cIn,
                 int M, int N, int K, int zdiv,
                 const float* __restrict__ sA, const float* __restrict__ sB,
                 float extra, const float* __restrict__ bias)
{
  __shared__ signed char LA[2*TILE_B];
  __shared__ signed char LB[2*TILE_B];
  int z = blockIdx.z;
  int zq = z / zdiv, zr = z % zdiv;
  A += (long long)zq*aOut + (long long)zr*aIn;
  B += (long long)zq*bOut + (long long)zr*bIn;
  C += (long long)zq*cOut + (long long)zr*cIn;

  int m0 = blockIdx.x * 128;
  int n0 = blockIdx.y * 128;
  int tid  = threadIdx.x;
  int lane = tid & 31, wave = tid >> 5;
  int wm = wave >> 1, wn = wave & 1;
  int hi = lane >> 4, mr = lane & 15;

  int fr = tid >> 1;
  int fc = (tid & 1) * 32;

  v8i acc[2][4] = {};
  int nsteps = K >> 6;
  bool interior = (m0 + 128 <= M) && (n0 + 128 <= N);

  if (interior){
    const signed char* ga = A + (long long)(m0 + fr)*lda + fc;
    const signed char* gb = B + (long long)(n0 + fr)*ldb + fc;
    unsigned la0 = (unsigned)(unsigned long long)&LA[fr*LDT + fc];
    unsigned lb0 = (unsigned)(unsigned long long)&LB[fr*LDT + fc];

    vit_async_b128(la0,      ga);
    vit_async_b128(la0 + 16, ga + 16);
    vit_async_b128(lb0,      gb);
    vit_async_b128(lb0 + 16, gb + 16);

    for (int t = 0; t < nsteps; ++t){
      int cur = t & 1;
      if (t + 1 < nsteps){
        int k1 = (t + 1) << 6;
        unsigned da = la0 + (cur ^ 1)*TILE_B;
        unsigned db = lb0 + (cur ^ 1)*TILE_B;
        vit_async_b128(da,      ga + k1);
        vit_async_b128(da + 16, ga + k1 + 16);
        vit_async_b128(db,      gb + k1);
        vit_async_b128(db + 16, gb + k1 + 16);
        vit_wait_async4();
      } else {
        vit_wait_async0();
      }
      __syncthreads();
      vit_mma8(LA + cur*TILE_B, LB + cur*TILE_B, wm, wn, hi, mr, acc);
      __syncthreads();
    }
  } else {
    for (int t = 0; t < nsteps; ++t){
      int k0 = t << 6;
      {
        int gm = m0 + fr;
        int4 v0 = make_int4(0,0,0,0), v1 = v0;
        if (gm < M){
          const signed char* p = A + (long long)gm*lda + k0 + fc;
          v0 = *(const int4*)(p);
          v1 = *(const int4*)(p + 16);
        }
        *(int4*)(&LA[fr*LDT + fc])      = v0;
        *(int4*)(&LA[fr*LDT + fc + 16]) = v1;
      }
      {
        int gn = n0 + fr;
        int4 v0 = make_int4(0,0,0,0), v1 = v0;
        if (gn < N){
          const signed char* p = B + (long long)gn*ldb + k0 + fc;
          v0 = *(const int4*)(p);
          v1 = *(const int4*)(p + 16);
        }
        *(int4*)(&LB[fr*LDT + fc])      = v0;
        *(int4*)(&LB[fr*LDT + fc + 16]) = v1;
      }
      __syncthreads();
      vit_mma8(LA, LB, wm, wn, hi, mr, acc);
      __syncthreads();
    }
  }

  float scale = sA[0] * sB[0] * extra;
  if (interior){
#pragma unroll
    for (int fm = 0; fm < 2; ++fm)
#pragma unroll
      for (int fn = 0; fn < 4; ++fn){
        int n = n0 + wn*64 + fn*16 + mr;
        float bv = bias ? bias[n] : 0.f;
#pragma unroll
        for (int r = 0; r < 8; ++r){
          int m = m0 + wm*32 + fm*16 + hi*8 + r;
          C[(long long)m*ldc + n] = (float)acc[fm][fn][r] * scale + bv;
        }
      }
  } else {
#pragma unroll
    for (int fm = 0; fm < 2; ++fm)
#pragma unroll
      for (int fn = 0; fn < 4; ++fn){
        int n = n0 + wn*64 + fn*16 + mr;
#pragma unroll
        for (int r = 0; r < 8; ++r){
          int m = m0 + wm*32 + fm*16 + hi*8 + r;
          if (m < M && n < N){
            float v = (float)acc[fm][fn][r] * scale;
            if (bias) v += bias[n];
            C[(long long)m*ldc + n] = v;
          }
        }
      }
  }
}

#define FLA_S 40
#define FLB_S 40

union HF16 { v16h v; unsigned int u[8]; };

__global__ __launch_bounds__(256)
void vit_gemm_av_f16(const __half* __restrict__ attnh, const signed char* __restrict__ qkvi8,
                     const float* __restrict__ sV, float* __restrict__ out)
{
  __shared__ __half Ah[128*FLA_S];
  __shared__ __half Bh[64*FLB_S];
  int z = blockIdx.z;
  int bIdx = z / HEADS, hIdx = z % HEADS;
  const __half* Abase = attnh + (long long)z * NTOK * KPAD;
  const signed char* vbase = qkvi8 + ((long long)bIdx*NTOK)*2304 + 1536 + hIdx*64;
  int m0 = blockIdx.x * 128;
  int tid  = threadIdx.x;
  int lane = tid & 31, wave = tid >> 5;
  int wm = wave >> 1, wn = wave & 1;
  int hi = lane >> 4, mr = lane & 15;

  v8f acc[2][2] = {};

  for (int k0 = 0; k0 < KPAD; k0 += 32){
    {
      int r = tid >> 1;
      int co = (tid & 1) * 16;
      int gm = m0 + r;
      int4 v0 = make_int4(0,0,0,0), v1 = v0;
      if (gm < NTOK){
        const int4* p = (const int4*)(Abase + (long long)gm*KPAD + k0 + co);
        v0 = p[0]; v1 = p[1];
      }
      __half* d = &Ah[r*FLA_S + co];
      *(int4*)d       = v0;
      *(int4*)(d + 8) = v1;
    }
    {
      int d = tid >> 2;
      int kk0 = (tid & 3) * 8;
#pragma unroll
      for (int j = 0; j < 8; ++j){
        int m = k0 + kk0 + j;
        float val = 0.f;
        if (m < NTOK) val = (float)vbase[(long long)m*2304 + d];
        Bh[d*FLB_S + kk0 + j] = __float2half(val);
      }
    }
    __syncthreads();

#pragma unroll
    for (int fm = 0; fm < 2; ++fm){
      const char* ap = (const char*)&Ah[(wm*32 + fm*16 + mr)*FLA_S];
      HF16 af;
#pragma unroll
      for (int i = 0; i < 8; ++i){
        int hoff = hi*8 + 2*i + ((i >> 2) << 3);
        af.u[i] = *(const unsigned int*)(ap + 2*hoff);
      }
#pragma unroll
      for (int fn = 0; fn < 2; ++fn){
        const char* bp = (const char*)&Bh[(wn*32 + fn*16 + mr)*FLB_S];
        HF16 bf;
#pragma unroll
        for (int i = 0; i < 8; ++i){
          bf.u[i] = *(const unsigned int*)(bp + hi*16 + 4*i + ((i >> 2) << 4));
        }
        acc[fm][fn] = __builtin_amdgcn_wmma_f32_16x16x32_f16(false, af.v, false, bf.v,
                                                             (short)0, acc[fm][fn], false, false);
      }
    }
    __syncthreads();
  }

  float sv = sV[0];
#pragma unroll
  for (int fm = 0; fm < 2; ++fm)
#pragma unroll
    for (int fn = 0; fn < 2; ++fn){
      int d = wn*32 + fn*16 + mr;
#pragma unroll
      for (int r = 0; r < 8; ++r){
        int m = m0 + wm*32 + fm*16 + hi*8 + r;
        if (m < NTOK)
          out[((long long)bIdx*NTOK + m)*DIMM + hIdx*64 + d] = acc[fm][fn][r] * sv;
      }
    }
}

static inline int gsz(long long n){
  long long b = (n + 255)/256;
  if (b > 4096) b = 4096;
  return (int)b;
}

extern "C" void kernel_launch(void* const* d_in, const int* in_sizes, int n_in,
                              void* d_out, int out_size, void* d_ws, size_t ws_size,
                              hipStream_t stream)
{
  const float* x       = (const float*)d_in[0];
  const float* conv_w  = (const float*)d_in[1];
  const float* conv_b  = (const float*)d_in[2];
  const float* cls_tok = (const float*)d_in[3];
  const float* pos_emb = (const float*)d_in[4];
  const float* qkv_w   = (const float*)d_in[5];
  const float* qkv_b   = (const float*)d_in[6];
  const float* proj_w  = (const float*)d_in[7];
  const float* proj_b  = (const float*)d_in[8];
  const float* ln1_w   = (const float*)d_in[9];
  const float* ln1_b   = (const float*)d_in[10];
  const float* ln2_w   = (const float*)d_in[11];
  const float* ln2_b   = (const float*)d_in[12];
  const float* fc1_w   = (const float*)d_in[13];
  const float* fc1_b   = (const float*)d_in[14];
  const float* fc2_w   = (const float*)d_in[15];
  const float* fc2_b   = (const float*)d_in[16];
  const float* norm_w  = (const float*)d_in[17];
  const float* norm_b  = (const float*)d_in[18];
  const float* head_w  = (const float*)d_in[19];
  const float* head_b  = (const float*)d_in[20];
  float* out = (float*)d_out;

  char* ws = (char*)d_ws;
  float*       SC    = (float*)(ws + O_SC);
  signed char* IM2   = (signed char*)(ws + O_IM2);
  signed char* WQKV  = (signed char*)(ws + O_WQKV);
  signed char* WPROJ = (signed char*)(ws + O_WPROJ);
  signed char* WFC1  = (signed char*)(ws + O_WFC1);
  signed char* WFC2  = (signed char*)(ws + O_WFC2);
  signed char* WHEAD = (signed char*)(ws + O_WHEAD);
  signed char* WCONV = (signed char*)(ws + O_WCONV);
  signed char* CLSI8 = (signed char*)(ws + O_CLSI8);
  float*       H     = (float*)(ws + O_H);
  float*       Abuf  = (float*)(ws + O_A);
  signed char* AI8   = (signed char*)(ws + O_AI8);
  float*       QKV   = (float*)(ws + O_QKV);
  signed char* QKVI8 = (signed char*)(ws + O_QKVI8);
  float*       ATTN  = (float*)(ws + O_ATTN);
  __half*      ATTNH = (__half*)(ws + O_ATTNH);
  float*       OBUF  = (float*)(ws + O_OBUF);
  float*       M1    = (float*)(ws + O_M1);
  signed char* M1I8  = (signed char*)(ws + O_M1I8);

  int slotIdx = 0;
  auto ctx = [&](){ float* p = SC + slotIdx; slotIdx += 8; return p; };

  auto absmax = [&](const float* p, long long n, float* s){
    vit_absmax<<<gsz(n), 256, 0, stream>>>(p, n, s);
  };
  auto fin = [&](float* s, float nl){
    vit_fin_scale<<<1, 1, 0, stream>>>(s, nl);
  };
  auto quant8 = [&](const float* p, signed char* q, long long n, const float* s){
    vit_quant_i8<<<gsz(n), 256, 0, stream>>>(p, q, n, s);
  };
  auto snap = [&](float* p, long long n, const float* s, float nl){
    vit_snap<<<gsz(n), 256, 0, stream>>>(p, n, s, nl);
  };
  auto ternarize = [&](const float* w, long long n, signed char* q, float* c){
    vit_sumabs<<<gsz(n), 256, 0, stream>>>(w, n, c);
    vit_ter_thr<<<1, 1, 0, stream>>>(c, (float)n);
    vit_ter_masked<<<gsz(n), 256, 0, stream>>>(w, n, c);
    vit_ter_alpha<<<1, 1, 0, stream>>>(c);
    vit_ter_write<<<gsz(n), 256, 0, stream>>>(w, q, n, c);
  };
  auto wquant8 = [&](const float* w, long long n, signed char* q, float* c){
    absmax(w, n, c); fin(c, 127.f); quant8(w, q, n, c+1);
  };
  auto gemm = [&](const signed char* A, int lda, long long aO, long long aI,
                  const signed char* B, int ldb, long long bO, long long bI,
                  float* C, int ldc, long long cO, long long cI,
                  int M, int N, int K, int Z, int zdiv,
                  const float* sA, const float* sB, float extra, const float* bias){
    dim3 g((M + 127)/128, (N + 127)/128, Z);
    vit_gemm_i8<<<g, 256, 0, stream>>>(A, lda, aO, aI, B, ldb, bO, bI, C, ldc, cO, cI,
                                       M, N, K, zdiv, sA, sB, extra, bias);
  };

  vit_zero_f32<<<8, 256, 0, stream>>>(SC, 2048);

  float* cx = ctx();
  absmax(x, 32LL*3*224*224, cx); fin(cx, 127.f);
  float* cw = ctx();
  wquant8(conv_w, 768LL*768, WCONV, cw);
  vit_im2col_q<<<gsz((long long)CONVTOK*DIMM), 256, 0, stream>>>(x, IM2, cx+1);
  gemm(IM2, DIMM, 0, 0, WCONV, DIMM, 0, 0, OBUF, DIMM, 0, 0,
       CONVTOK, DIMM, DIMM, 1, 1, cx+1, cw+1, 1.f, conv_b);
  float* c16c = ctx();
  absmax(OBUF, (long long)CONVTOK*DIMM, c16c); fin(c16c, 32767.f);
  snap(OBUF, (long long)CONVTOK*DIMM, c16c+1, 32767.f);

  float* cp = ctx();
  absmax(pos_emb, (long long)NTOK*DIMM, cp); fin(cp, 127.f);
  vit_build_h<<<gsz((long long)TOK*DIMM), 256, 0, stream>>>(H, cls_tok, pos_emb, OBUF, cp+1);
  float* ch0 = ctx();
  absmax(H, (long long)TOK*DIMM, ch0); fin(ch0, 32767.f);
  snap(H, (long long)TOK*DIMM, ch0+1, 32767.f);

  for (int i = 0; i < 12; ++i){
    const float* Wqkv = qkv_w + (long long)i*2304*768;
    const float* Bqkv = qkv_b + (long long)i*2304;
    const float* Wprj = proj_w + (long long)i*768*768;
    const float* Bprj = proj_b + (long long)i*768;
    const float* Wf1  = fc1_w + (long long)i*3072*768;
    const float* Bf1  = fc1_b + (long long)i*3072;
    const float* Wf2  = fc2_w + (long long)i*768*3072;
    const float* Bf2  = fc2_b + (long long)i*768;

    vit_ln<<<(TOK + 7)/8, 256, 0, stream>>>(H, ln1_w + i*768, ln1_b + i*768, Abuf, TOK);
    float* ca = ctx();
    absmax(Abuf, (long long)TOK*DIMM, ca); fin(ca, 127.f);
    quant8(Abuf, AI8, (long long)TOK*DIMM, ca+1);
    float* ct1 = ctx();
    ternarize(Wqkv, 2304LL*768, WQKV, ct1);
    gemm(AI8, DIMM, 0, 0, WQKV, DIMM, 0, 0, QKV, 2304, 0, 0,
         TOK, 2304, DIMM, 1, 1, ca+1, ct1+4, 1.f, Bqkv);
    float* cq = ctx();
    absmax(QKV, (long long)TOK*2304, cq); fin(cq, 127.f);
    quant8(QKV, QKVI8, (long long)TOK*2304, cq+1);

    gemm(QKVI8,       2304, (long long)NTOK*2304, 64,
         QKVI8 + 768, 2304, (long long)NTOK*2304, 64,
         ATTN,  NTOK, 12LL*NTOK*NTOK, (long long)NTOK*NTOK,
         NTOK, NTOK, HD, ZBH, HEADS, cq+1, cq+1, 0.125f, nullptr);
    float* cat = ctx();
    absmax(ATTN, (long long)ZBH*NTOK*NTOK, cat); fin(cat, 127.f);
    vit_softmax<<<((long long)ZBH*NTOK + 7)/8, 256, 0, stream>>>(ATTN, ATTNH, cat+1);

    {
      dim3 g(2, 1, ZBH);
      vit_gemm_av_f16<<<g, 256, 0, stream>>>(ATTNH, QKVI8, cq+1, OBUF);
    }
    float* co = ctx();
    absmax(OBUF, (long long)TOK*DIMM, co); fin(co, 127.f);
    quant8(OBUF, AI8, (long long)TOK*DIMM, co+1);
    float* cpw = ctx();
    wquant8(Wprj, 768LL*768, WPROJ, cpw);
    gemm(AI8, DIMM, 0, 0, WPROJ, DIMM, 0, 0, Abuf, DIMM, 0, 0,
         TOK, DIMM, DIMM, 1, 1, co+1, cpw+1, 1.f, Bprj);
    float* c16 = ctx();
    absmax(Abuf, (long long)TOK*DIMM, c16); fin(c16, 32767.f);
    snap(Abuf, (long long)TOK*DIMM, c16+1, 32767.f);
    vit_add<<<gsz((long long)TOK*DIMM), 256, 0, stream>>>(H, Abuf, (long long)TOK*DIMM);
    float* chA = ctx();
    absmax(H, (long long)TOK*DIMM, chA); fin(chA, 32767.f);
    snap(H, (long long)TOK*DIMM, chA+1, 32767.f);

    vit_ln<<<(TOK + 7)/8, 256, 0, stream>>>(H, ln2_w + i*768, ln2_b + i*768, Abuf, TOK);
    float* cm = ctx();
    absmax(Abuf, (long long)TOK*DIMM, cm); fin(cm, 127.f);
    quant8(Abuf, AI8, (long long)TOK*DIMM, cm+1);
    float* ct2 = ctx();
    ternarize(Wf1, 3072LL*768, WFC1, ct2);
    gemm(AI8, DIMM, 0, 0, WFC1, DIMM, 0, 0, M1, 3072, 0, 0,
         TOK, 3072, DIMM, 1, 1, cm+1, ct2+4, 1.f, Bf1);
    float* c8a = ctx();
    absmax(M1, (long long)TOK*3072, c8a); fin(c8a, 127.f);
    snap(M1, (long long)TOK*3072, c8a+1, 127.f);
    vit_gelu<<<gsz((long long)TOK*3072), 256, 0, stream>>>(M1, (long long)TOK*3072);
    float* c8b = ctx();
    absmax(M1, (long long)TOK*3072, c8b); fin(c8b, 127.f);
    quant8(M1, M1I8, (long long)TOK*3072, c8b+1);
    float* ct3 = ctx();
    ternarize(Wf2, 768LL*3072, WFC2, ct3);
    gemm(M1I8, 3072, 0, 0, WFC2, 3072, 0, 0, Abuf, DIMM, 0, 0,
         TOK, DIMM, 3072, 1, 1, c8b+1, ct3+4, 1.f, Bf2);
    float* c16b = ctx();
    absmax(Abuf, (long long)TOK*DIMM, c16b); fin(c16b, 32767.f);
    snap(Abuf, (long long)TOK*DIMM, c16b+1, 32767.f);
    vit_add<<<gsz((long long)TOK*DIMM), 256, 0, stream>>>(H, Abuf, (long long)TOK*DIMM);
    float* chB = ctx();
    absmax(H, (long long)TOK*DIMM, chB); fin(chB, 32767.f);
    snap(H, (long long)TOK*DIMM, chB+1, 32767.f);
  }

  vit_ln<<<(TOK + 7)/8, 256, 0, stream>>>(H, norm_w, norm_b, Abuf, TOK);
  float* ccls = ctx();
  vit_absmax_cls<<<(BATCH*DIMM + 255)/256, 256, 0, stream>>>(Abuf, ccls);
  fin(ccls, 127.f);
  vit_quant_cls<<<(BATCH*DIMM + 255)/256, 256, 0, stream>>>(Abuf, CLSI8, ccls+1);
  float* chw = ctx();
  wquant8(head_w, 1000LL*768, WHEAD, chw);
  gemm(CLSI8, DIMM, 0, 0, WHEAD, DIMM, 0, 0, out, 1000, 0, 0,
       BATCH, 1000, DIMM, 1, 1, ccls+1, chw+1, 1.f, head_b);

  (void)in_sizes; (void)n_in; (void)out_size; (void)ws_size;
}
